// NormalEncorder_7834020348450
// MI455X (gfx1250) — matrix operand order repaired and checked by reading, not compiled, not run on hardware
//
#include <hip/hip_runtime.h>
#include <stdint.h>

// ---------------------------------------------------------------------------
// NormalEncoder forward for MI455X (gfx1250, wave32, WMMA bf16 16x16x32).
// B=4, C=128, N=4096, k=8.  All heavy GEMMs (c11, Gram matrices for kNN,
// stacked edge-convs, c22) run on v_wmma_f32_16x16x32_bf16 with f32 accum.
// EdgeConv is algebraically collapsed to GEMM + gather-max (linearity + the
// monotonicity of BN(+gamma)+LeakyReLU lets max_k commute with activation).
// ---------------------------------------------------------------------------

#define BATCH 4
#define NPTS  4096
#define NB    (BATCH * NPTS)
#define EPSV  1e-5f

typedef __attribute__((ext_vector_type(16))) __bf16 v16bf;
typedef __attribute__((ext_vector_type(8)))  float  v8f;

union Frag { v16bf v; uint4 u[2]; };

__device__ __forceinline__ unsigned short f2bf(float f) {
    union { float f; unsigned u; } x; x.f = f;
    unsigned r = x.u + 0x7FFFu + ((x.u >> 16) & 1u);   // round-to-nearest-even
    return (unsigned short)(r >> 16);
}
__device__ __forceinline__ float bf2f(unsigned short h) {
    union { unsigned u; float f; } x; x.u = ((unsigned)h) << 16;
    return x.f;
}
__device__ __forceinline__ float lrelu(float v) { return v >= 0.f ? v : 0.2f * v; }

// A fragment: 16x32 bf16 tile, row-major src [16][ld], lane m holds row m.
// lanes 0-15: v0..3 = K 0..7, v4..7 = K 16..23 ; lanes 16-31: +8.
__device__ __forceinline__ Frag loadA(const unsigned short* p0, int ld) {
    int lane = threadIdx.x & 31;
    const unsigned short* p = p0 + (lane & 15) * ld + ((lane >> 4) << 3);
    Frag f;
    f.u[0] = *(const uint4*)(p);
    f.u[1] = *(const uint4*)(p + 16);
    return f;
}
// B fragment: 32x16 bf16, sourced from TRANSPOSED array Bt[n][k] (row n = col).
// lanes 0-15: v0..3 = K 0..7, v4..7 = K 16..23 ; lanes 16-31: +8.
__device__ __forceinline__ Frag loadB(const unsigned short* p0, int ld) {
    int lane = threadIdx.x & 31;
    const unsigned short* p = p0 + (lane & 15) * ld + ((lane >> 4) << 3);
    Frag f;
    f.u[0] = *(const uint4*)(p);
    f.u[1] = *(const uint4*)(p + 16);
    return f;
}
__device__ __forceinline__ v8f wmma_bf(const Frag& a, const Frag& b, v8f c) {
    return __builtin_amdgcn_wmma_f32_16x16x32_bf16(false, a.v, false, b.v,
                                                   (short)0, c, false, false);
}
// C tile (16x16 f32): element r -> row  (r + ((lane>>4)<<3)), col (lane&15).
__device__ __forceinline__ v8f gemm_tile(const unsigned short* Arow,
                                         const unsigned short* Bt, int K) {
    v8f acc = {};
    for (int k0 = 0; k0 < K; k0 += 32) {
        if (k0 + 32 < K) __builtin_prefetch((const void*)(Bt + k0 + 32), 0, 1);
        Frag a = loadA(Arow + k0, K);
        Frag b = loadB(Bt + k0, K);
        acc = wmma_bf(a, b, acc);
    }
    return acc;
}

// ---------------- small prep kernels -----------------------------------

// fold conv bias + BN into scale/shift: act(in) = lrelu(scale*gemm + shift)
__global__ void k_bnprep(const float* bias, const float* g, const float* be,
                         const float* mean, const float* var,
                         float* sc, float* sh, int M) {
    int t = blockIdx.x * blockDim.x + threadIdx.x;
    if (t >= M) return;
    float s = g[t] * rsqrtf(var[t] + EPSV);
    sc[t] = s;
    sh[t] = s * (bias[t] - mean[t]) + be[t];
}

__global__ void k_f2bf(const float* in, unsigned short* out, int n) {
    int t = blockIdx.x * blockDim.x + threadIdx.x;
    if (t < n) out[t] = f2bf(in[t]);
}
// stack c12 W [64][256] -> [128][128]: rows 0..63 = W1(=W[:,:128]), 64..127 = W2
__global__ void k_stack12(const float* in, unsigned short* out) {
    int t = blockIdx.x * blockDim.x + threadIdx.x;
    if (t >= 128 * 128) return;
    int o = t >> 7, c = t & 127;
    float v = (o < 64) ? in[o * 256 + c] : in[(o - 64) * 256 + 128 + c];
    out[t] = f2bf(v);
}
// stack c21 W [128][128] -> [256][64]
__global__ void k_stack21(const float* in, unsigned short* out) {
    int t = blockIdx.x * blockDim.x + threadIdx.x;
    if (t >= 256 * 64) return;
    int o = t >> 6, c = t & 63;
    float v = (o < 128) ? in[o * 128 + c] : in[(o - 128) * 128 + 64 + c];
    out[t] = f2bf(v);
}
// h = concat(x+nf, pff)  stored transposed bf16:  ht[b][n][c], c in 0..255
__global__ void k_build_ht(const float* x, const float* nf, const float* pff,
                           unsigned short* ht) {
    int t = blockIdx.x * blockDim.x + threadIdx.x;   // t = ((b*N)+n)*256 + c
    if (t >= NB * 256) return;
    int c = t & 255, n = (t >> 8) & (NPTS - 1), b = t >> 20;
    float v;
    if (c < 128) {
        int i = (b * 128 + c) * NPTS + n;
        v = x[i] + nf[i];
    } else {
        v = pff[(b * 128 + (c - 128)) * NPTS + n];
    }
    ht[t] = f2bf(v);
}
// column squared norms of transposed bf16 features [NB][K]
__global__ void k_colnorm(const unsigned short* xt, float* cn, int K) {
    int t = blockIdx.x * blockDim.x + threadIdx.x;
    if (t >= NB) return;
    const unsigned short* p = xt + t * K;
    float s = 0.f;
    for (int c = 0; c < K; ++c) { float v = bf2f(p[c]); s = fmaf(v, v, s); }
    cn[t] = s;
}

// ---------------- WMMA GEMM kernels ------------------------------------
// grid: (N/16, M/128, B), block 256 = 8 waves; wave w owns rows m0..m0+15.

__global__ void k_gemm_bn_act(const unsigned short* Wb, const unsigned short* Xt,
                              const float* sc, const float* sh,
                              float* fT, unsigned short* ftb, int K) {
    int b = blockIdx.z, n0 = blockIdx.x * 16;
    int w = threadIdx.x >> 5, lane = threadIdx.x & 31;
    int m0 = blockIdx.y * 128 + w * 16;
    v8f acc = gemm_tile(Wb + m0 * K, Xt + (b * NPTS + n0) * K, K);
    int n = n0 + (lane & 15);
    int mb = m0 + ((lane >> 4) << 3);
#pragma unroll
    for (int r = 0; r < 8; ++r) {
        int m = mb + r;
        float v = lrelu(fmaf(acc[r], sc[m], sh[m]));
        int o = (b * NPTS + n) * 128 + m;
        fT[o] = v;
        ftb[o] = f2bf(v);
    }
}

__global__ void k_gemm_plain(const unsigned short* Wb, const unsigned short* Xt,
                             float* Pt, int K, int ldout) {
    int b = blockIdx.z, n0 = blockIdx.x * 16;
    int w = threadIdx.x >> 5, lane = threadIdx.x & 31;
    int m0 = blockIdx.y * 128 + w * 16;
    v8f acc = gemm_tile(Wb + m0 * K, Xt + (b * NPTS + n0) * K, K);
    int n = n0 + (lane & 15);
    int mb = m0 + ((lane >> 4) << 3);
#pragma unroll
    for (int r = 0; r < 8; ++r)
        Pt[(b * NPTS + n) * ldout + mb + r] = acc[r];
}

// c22 GEMM + act + per-tile column max  -> pmax[b][m][ntile]
__global__ void k_gemm_colmax(const unsigned short* Wb, const unsigned short* Xt,
                              const float* sc, const float* sh,
                              float* pmax, int K) {
    int b = blockIdx.z, n0 = blockIdx.x * 16;
    int w = threadIdx.x >> 5, lane = threadIdx.x & 31;
    int m0 = blockIdx.y * 128 + w * 16;
    v8f acc = gemm_tile(Wb + m0 * K, Xt + (b * NPTS + n0) * K, K);
    int mb = m0 + ((lane >> 4) << 3);
#pragma unroll
    for (int r = 0; r < 8; ++r) {
        int m = mb + r;
        float v = lrelu(fmaf(acc[r], sc[m], sh[m]));
        v = fmaxf(v, __shfl_xor(v, 1, 32));
        v = fmaxf(v, __shfl_xor(v, 2, 32));
        v = fmaxf(v, __shfl_xor(v, 4, 32));
        v = fmaxf(v, __shfl_xor(v, 8, 32));
        if ((lane & 15) == 0)
            pmax[(b * 256 + m) * 256 + blockIdx.x] = v;
    }
}

// ---------------- fused Gram + top-8 (kNN) ------------------------------
// S[n,m] = 2<f_n,f_m> - |f_m|^2 (row-constant -|f_n|^2 dropped: argmax-safe).
// One block = 16 rows; 8 waves stride 256 column tiles; top-8 kept in regs.

template <int KD>
__global__ void k_topk(const unsigned short* xt, const float* cn, int* idxout) {
    __shared__ float slab[8][16][17];
    __shared__ float mv[8][16][8];
    __shared__ int   mi[8][16][8];
    int b = blockIdx.y, r0 = blockIdx.x * 16;
    int w = threadIdx.x >> 5, lane = threadIdx.x & 31;
    const unsigned short* xb = xt + b * NPTS * KD;

    Frag af[KD / 32];
#pragma unroll
    for (int kt = 0; kt < KD / 32; ++kt)
        af[kt] = loadA(xb + r0 * KD + kt * 32, KD);

    float tv[8]; int ti[8];
#pragma unroll
    for (int j = 0; j < 8; ++j) { tv[j] = -3.4e38f; ti[j] = 0; }

    for (int ct = w; ct < NPTS / 16; ct += 8) {      // 32 iters, uniform
        int c0 = ct * 16;
        v8f acc = {};
#pragma unroll
        for (int kt = 0; kt < KD / 32; ++kt) {
            Frag bf = loadB(xb + c0 * KD + kt * 32, KD);
            acc = wmma_bf(af[kt], bf, acc);
        }
        int nn = lane & 15;
        float cnv = cn[b * NPTS + c0 + nn];
        int mb = (lane >> 4) << 3;
#pragma unroll
        for (int r = 0; r < 8; ++r)
            slab[w][mb + r][nn] = 2.0f * acc[r] - cnv;
        __syncthreads();
        if (lane < 16) {
#pragma unroll
            for (int j = 0; j < 16; ++j) {
                float v = slab[w][lane][j];
                int id = c0 + j;
                if (v > tv[7]) {
#pragma unroll
                    for (int p = 0; p < 8; ++p)
                        if (v > tv[p]) {
                            float t0 = tv[p]; tv[p] = v; v = t0;
                            int   t1 = ti[p]; ti[p] = id; id = t1;
                        }
                }
            }
        }
        __syncthreads();
    }
    if (lane < 16) {
#pragma unroll
        for (int j = 0; j < 8; ++j) { mv[w][lane][j] = tv[j]; mi[w][lane][j] = ti[j]; }
    }
    __syncthreads();
    if (threadIdx.x < 16) {
        int row = threadIdx.x;
        float fv[8]; int fi[8];
#pragma unroll
        for (int j = 0; j < 8; ++j) { fv[j] = -3.4e38f; fi[j] = 0; }
        for (int ww = 0; ww < 8; ++ww)
#pragma unroll
            for (int j = 0; j < 8; ++j) {
                float v = mv[ww][row][j]; int id = mi[ww][row][j];
                if (v > fv[7]) {
#pragma unroll
                    for (int p = 0; p < 8; ++p)
                        if (v > fv[p]) {
                            float t0 = fv[p]; fv[p] = v; v = t0;
                            int   t1 = fi[p]; fi[p] = id; id = t1;
                        }
                }
            }
#pragma unroll
        for (int j = 0; j < 8; ++j)
            idxout[(b * NPTS + r0 + row) * 8 + j] = fi[j];
    }
}

// ---------------- gather-max epilogue of collapsed EdgeConv -------------
// out[n,o] = act( max_k P[idx[n,k]][o] + P[n][CO+o] - P[n][o] ) (+ addT)
__global__ void k_gather(const float* Pt, const int* idx,
                         const float* sc, const float* sh,
                         const float* addT, unsigned short* outb,
                         int CO, int coShift) {
    int t = blockIdx.x * blockDim.x + threadIdx.x;   // t = gn*CO + o
    int o = t & (CO - 1);
    int gn = t >> coShift;
    if (gn >= NB) return;
    int b = gn >> 12;
    int ld = CO * 2;
    const int* ir = idx + gn * 8;
    float mx = -3.4e38f;
#pragma unroll
    for (int k = 0; k < 8; ++k) {
        int j = ir[k];
        mx = fmaxf(mx, Pt[(b * NPTS + j) * ld + o]);
    }
    float raw = mx + Pt[gn * ld + CO + o] - Pt[gn * ld + o];
    float v = lrelu(fmaf(raw, sc[o], sh[o]));
    if (addT) v += addT[gn * CO + o];
    outb[gn * CO + o] = f2bf(v);
}

// ---------------- FC head ----------------------------------------------
__global__ void k_head(const float* pmax,
                       const float* f1W, const float* scf1, const float* shf1,
                       const float* f2W, const float* scf2, const float* shf2,
                       const float* f3W, const float* f3b, float* out) {
    __shared__ float dbuf[256], e1[128], e2[64];
    int b = blockIdx.x, tid = threadIdx.x;
    {
        float v = -3.4e38f;
        const float* pm = pmax + (b * 256 + tid) * 256;
        for (int t = 0; t < 256; ++t) v = fmaxf(v, pm[t]);
        dbuf[tid] = v;
    }
    __syncthreads();
    if (tid < 128) {
        const float* wr = f1W + tid * 256;
        float s = 0.f;
        for (int c = 0; c < 256; ++c) s = fmaf(wr[c], dbuf[c], s);
        e1[tid] = lrelu(fmaf(s, scf1[tid], shf1[tid]));
    }
    __syncthreads();
    if (tid < 64) {
        const float* wr = f2W + tid * 128;
        float s = 0.f;
        for (int c = 0; c < 128; ++c) s = fmaf(wr[c], e1[c], s);
        e2[tid] = lrelu(fmaf(s, scf2[tid], shf2[tid]));
    }
    __syncthreads();
    if (tid < 3) {
        const float* wr = f3W + tid * 64;
        float s = f3b[tid];
        for (int c = 0; c < 64; ++c) s = fmaf(wr[c], e2[c], s);
        out[b * 3 + tid] = s;
    }
}

// ---------------- host orchestration ------------------------------------
extern "C" void kernel_launch(void* const* d_in, const int* in_sizes, int n_in,
                              void* d_out, int out_size, void* d_ws, size_t ws_size,
                              hipStream_t stream) {
    // input order: x, nf, pff, then per block {W,b,g,be,m,v}:
    // c11=3..8, c12=9..14, c21=15..20, c22=21..26, f1=27..32, f2=33..38, f3W=39,f3b=40
    const float* x   = (const float*)d_in[0];
    const float* nf  = (const float*)d_in[1];
    const float* pff = (const float*)d_in[2];
    auto F = [&](int i) { return (const float*)d_in[i]; };

    char* ws = (char*)d_ws;
    size_t off = 0;
    auto alloc = [&](size_t bytes) { size_t o = off; off += (bytes + 255) & ~(size_t)255; return o; };

    unsigned short* ht    = (unsigned short*)(ws + alloc((size_t)NB * 256 * 2));
    unsigned short* wc11b = (unsigned short*)(ws + alloc(128 * 256 * 2));
    unsigned short* ws12b = (unsigned short*)(ws + alloc(128 * 128 * 2));
    unsigned short* ws21b = (unsigned short*)(ws + alloc(256 * 64 * 2));
    unsigned short* wc22b = (unsigned short*)(ws + alloc(256 * 128 * 2));
    float* scales = (float*)(ws + alloc(768 * 4));
    float* shifts = (float*)(ws + alloc(768 * 4));
    float*          fT   = (float*)(ws + alloc((size_t)NB * 128 * 4));
    unsigned short* ftb  = (unsigned short*)(ws + alloc((size_t)NB * 128 * 2));
    float*          cn1  = (float*)(ws + alloc((size_t)NB * 4));
    int*            idx1 = (int*)(ws + alloc((size_t)NB * 8 * 4));
    float*          P1t  = (float*)(ws + alloc((size_t)NB * 128 * 4));
    unsigned short* f1tb = (unsigned short*)(ws + alloc((size_t)NB * 64 * 2));
    float*          cn2  = (float*)(ws + alloc((size_t)NB * 4));
    int*            idx2 = (int*)(ws + alloc((size_t)NB * 8 * 4));
    float*          P2t  = (float*)(ws + alloc((size_t)NB * 256 * 4));
    unsigned short* gtb  = (unsigned short*)(ws + alloc((size_t)NB * 128 * 2));
    float*          pmax = (float*)(ws + alloc((size_t)BATCH * 256 * 256 * 4));

    float* sc11 = scales + 0,   * sh11 = shifts + 0;    // 128
    float* sc12 = scales + 128, * sh12 = shifts + 128;  // 64
    float* sc21 = scales + 192, * sh21 = shifts + 192;  // 128
    float* sc22 = scales + 320, * sh22 = shifts + 320;  // 256
    float* scf1 = scales + 576, * shf1 = shifts + 576;  // 128
    float* scf2 = scales + 704, * shf2 = shifts + 704;  // 64

    // BN folds
    k_bnprep<<<1, 256, 0, stream>>>(F(4),  F(5),  F(6),  F(7),  F(8),  sc11, sh11, 128);
    k_bnprep<<<1, 256, 0, stream>>>(F(10), F(11), F(12), F(13), F(14), sc12, sh12, 64);
    k_bnprep<<<1, 256, 0, stream>>>(F(16), F(17), F(18), F(19), F(20), sc21, sh21, 128);
    k_bnprep<<<1, 256, 0, stream>>>(F(22), F(23), F(24), F(25), F(26), sc22, sh22, 256);
    k_bnprep<<<1, 256, 0, stream>>>(F(28), F(29), F(30), F(31), F(32), scf1, shf1, 128);
    k_bnprep<<<1, 256, 0, stream>>>(F(34), F(35), F(36), F(37), F(38), scf2, shf2, 64);

    // weights -> bf16 (+ stacked forms for collapsed EdgeConvs)
    k_f2bf   <<<128, 256, 0, stream>>>(F(3),  wc11b, 128 * 256);
    k_stack12<<<64,  256, 0, stream>>>(F(9),  ws12b);
    k_stack21<<<64,  256, 0, stream>>>(F(15), ws21b);
    k_f2bf   <<<128, 256, 0, stream>>>(F(21), wc22b, 256 * 128);

    // h^T bf16
    k_build_ht<<<NB, 256, 0, stream>>>(x, nf, pff, ht);

    // c11: f = act(W*h)   (store f32 transposed + bf16 transposed)
    k_gemm_bn_act<<<dim3(256, 1, BATCH), 256, 0, stream>>>(wc11b, ht, sc11, sh11, fT, ftb, 256);

    // kNN #1 on f (K=128)
    k_colnorm<<<NB / 256, 256, 0, stream>>>(ftb, cn1, 128);
    k_topk<128><<<dim3(256, BATCH), 256, 0, stream>>>(ftb, cn1, idx1);

    // EdgeConv #1 collapsed: P1 = [W1;W2]*f, then gather-max + act -> f1
    k_gemm_plain<<<dim3(256, 1, BATCH), 256, 0, stream>>>(ws12b, ftb, P1t, 128, 128);
    k_gather<<<(NB * 64) / 256, 256, 0, stream>>>(P1t, idx1, sc12, sh12, nullptr, f1tb, 64, 6);

    // kNN #2 on f1 (K=64)
    k_colnorm<<<NB / 256, 256, 0, stream>>>(f1tb, cn2, 64);
    k_topk<64><<<dim3(256, BATCH), 256, 0, stream>>>(f1tb, cn2, idx2);

    // EdgeConv #2 collapsed + residual add of f -> g (bf16)
    k_gemm_plain<<<dim3(256, 2, BATCH), 256, 0, stream>>>(ws21b, f1tb, P2t, 64, 256);
    k_gather<<<(NB * 128) / 256, 256, 0, stream>>>(P2t, idx2, sc21, sh21, fT, gtb, 128, 7);

    // c22 + act + per-tile column max
    k_gemm_colmax<<<dim3(256, 2, BATCH), 256, 0, stream>>>(wc22b, gtb, sc22, sh22, pmax, 128);

    // global max + FC head
    k_head<<<BATCH, 256, 0, stream>>>(pmax, F(27), scf1, shf1, F(33), scf2, shf2,
                                      F(39), F(40), (float*)d_out);
    (void)in_sizes; (void)n_in; (void)out_size; (void)ws_size;
}
